// HierarchicalQuantilePredictor_85693187490440
// MI455X (gfx1250) — compile-verified, matrix operand order checked by reading, not run on hardware
//
#include <hip/hip_runtime.h>
#include <math.h>

typedef __attribute__((ext_vector_type(16))) __bf16 v16bf;
typedef __attribute__((ext_vector_type(8)))  float  v8f;
typedef __attribute__((ext_vector_type(4)))  unsigned int u32x4;

#define B_ROWS 8192
#define HID    1024
#define NHEAD  64
#define NQ     99
#define NINC   640
#define NTOT   704   /* 64 median cols + 640 inc cols */

// ---------------- conversion: fp32 -> bf16 hi/lo split ----------------
__global__ __launch_bounds__(256) void cvt_x_kernel(const float* __restrict__ x,
                                                    __bf16* __restrict__ xh,
                                                    __bf16* __restrict__ xl, int n) {
  int i = blockIdx.x * 256 + threadIdx.x;
  if (i >= n) return;
  float v = x[i];
  __bf16 h = (__bf16)v;
  xh[i] = h;
  xl[i] = (__bf16)(v - (float)h);
}

// Fuse [Wm | Wi] and transpose to (NTOT x HID) row-major so B-fragment
// loads are contiguous in K (same addressing as A).
__global__ __launch_bounds__(256) void cvt_w_kernel(const float* __restrict__ Wm,
                                                    const float* __restrict__ Wi,
                                                    __bf16* __restrict__ wth,
                                                    __bf16* __restrict__ wtl) {
  int i = blockIdx.x * 256 + threadIdx.x;   // over NTOT*HID
  if (i >= NTOT * HID) return;
  int n = i / HID, k = i % HID;
  float v = (n < NHEAD) ? Wm[k * NHEAD + n] : Wi[k * NINC + (n - NHEAD)];
  __bf16 h = (__bf16)v;
  wth[i] = h;
  wtl[i] = (__bf16)(v - (float)h);
}

// ---------------- WMMA GEMM ----------------
// Fragment layout (16-bit A 16x32 / B 32x16, wave32):
//   lane L: row/col = L%16, half = L/16
//   element e (0..15): K = (e/8)*16 + half*8 + (e%8)
// => elements 0..7 contiguous at (k0 + half*8), elements 8..15 at +16.
__device__ __forceinline__ v16bf load_frag(const __bf16* p) {
  union U { u32x4 u[2]; v16bf v; };
  U t;
  t.u[0] = *(const u32x4*)(p);
  t.u[1] = *(const u32x4*)(p + 16);
  return t.v;
}

__device__ __forceinline__ float softplus_f(float v) {
  return (v > 20.0f) ? v : log1pf(expf(v));
}

// grid (8192/256, 704/64), block 256 (8 waves).
// Wave tile: 64M x 32N = 4x2 accumulators of 16x16.
__global__ __launch_bounds__(256) void gemm_kernel(const __bf16* __restrict__ xh,
                                                   const __bf16* __restrict__ xl,
                                                   const __bf16* __restrict__ wth,
                                                   const __bf16* __restrict__ wtl,
                                                   const float* __restrict__ bm,
                                                   const float* __restrict__ bi,
                                                   float* __restrict__ pre) {
  const int lane  = threadIdx.x & 31;
  const int wave  = threadIdx.x >> 5;
  const int mwave = wave >> 1;            // 0..3
  const int nwave = wave & 1;             // 0..1
  const int half  = lane >> 4;            // 0/1
  const int lrow  = lane & 15;            // row of A / col of B / col of D
  const int m_base = blockIdx.x * 256 + mwave * 64;
  const int n_base = blockIdx.y * 64  + nwave * 32;

  v8f acc[4][2];
#pragma unroll
  for (int mt = 0; mt < 4; ++mt)
#pragma unroll
    for (int nt = 0; nt < 2; ++nt)
      acc[mt][nt] = v8f{};

  size_t arow[4], brow[2];
#pragma unroll
  for (int mt = 0; mt < 4; ++mt)
    arow[mt] = (size_t)(m_base + mt * 16 + lrow) * HID + (size_t)(half * 8);
#pragma unroll
  for (int nt = 0; nt < 2; ++nt)
    brow[nt] = (size_t)(n_base + nt * 16 + lrow) * HID + (size_t)(half * 8);

  for (int k0 = 0; k0 < HID; k0 += 32) {
    v16bf ah[4], al[4], bh[2], bl[2];
#pragma unroll
    for (int mt = 0; mt < 4; ++mt) {
      ah[mt] = load_frag(xh + arow[mt] + k0);
      al[mt] = load_frag(xl + arow[mt] + k0);
    }
#pragma unroll
    for (int nt = 0; nt < 2; ++nt) {
      bh[nt] = load_frag(wth + brow[nt] + k0);
      bl[nt] = load_frag(wtl + brow[nt] + k0);
    }
    // 3-term bf16 split: hi*hi + hi*lo + lo*hi  (~fp32 accuracy)
#pragma unroll
    for (int mt = 0; mt < 4; ++mt)
#pragma unroll
      for (int nt = 0; nt < 2; ++nt) {
        v8f c = acc[mt][nt];
        c = __builtin_amdgcn_wmma_f32_16x16x32_bf16(false, ah[mt], false, bh[nt], (short)0, c, false, false);
        c = __builtin_amdgcn_wmma_f32_16x16x32_bf16(false, ah[mt], false, bl[nt], (short)0, c, false, false);
        c = __builtin_amdgcn_wmma_f32_16x16x32_bf16(false, al[mt], false, bh[nt], (short)0, c, false, false);
        acc[mt][nt] = c;
      }
  }

  // Epilogue: D element r -> (m = base + r + 8*half, n = base + lane%16)
#pragma unroll
  for (int mt = 0; mt < 4; ++mt)
#pragma unroll
    for (int nt = 0; nt < 2; ++nt) {
      int n = n_base + nt * 16 + lrow;
      float bias = (n < NHEAD) ? bm[n] : bi[n - NHEAD];
      bool is_inc = (n >= NHEAD);
#pragma unroll
      for (int r = 0; r < 8; ++r) {
        int m = m_base + mt * 16 + r + 8 * half;
        float v = acc[mt][nt][r] + bias;
        if (is_inc) v = softplus_f(v);
        pre[(size_t)m * NTOT + n] = v;
      }
    }
}

// ---------------- quantile expansion ----------------
// One block per batch row. pattern is periodic (period 10):
//   cum0[s] = (s/10)*c10 + P[s%10],  P = prefix sums of inc[0..9].
__global__ __launch_bounds__(256) void quant_kernel(const float* __restrict__ quantiles,
                                                    const float* __restrict__ pre,
                                                    float* __restrict__ out) {
  __shared__ float sq[NQ];
  __shared__ float sortedq[NQ];
  __shared__ int   inv[NQ];      // inv[j] = rank of q_j  (== argsort(sort_idx)[j])
  __shared__ int   midx;

  const int b = blockIdx.x;
  const int t = threadIdx.x;
  const float* q = quantiles + (size_t)b * NQ;

  if (t < NQ) sq[t] = q[t];
  __syncthreads();

  if (t < NQ) {
    float v = sq[t];
    int r = 0;
    for (int i = 0; i < NQ; ++i) {
      float u = sq[i];
      r += (u < v) || (u == v && i < t);   // stable rank == argsort semantics
    }
    inv[t] = r;
    sortedq[r] = v;
  }
  __syncthreads();

  if (t == 0) {
    int best = 0;
    float bd = fabsf(sortedq[0] - 0.5f);
    for (int r = 1; r < NQ; ++r) {
      float d = fabsf(sortedq[r] - 0.5f);
      if (d < bd) { bd = d; best = r; }    // first-index tie-break like argmin
    }
    midx = best;
  }
  __syncthreads();

  const int mi = midx;
  const int h  = t >> 2;        // 64 heads, 4 threads each
  const int j0 = t & 3;

  const float* rowp = pre + (size_t)b * NTOT;
  const float med = rowp[h];
  float P[11];
  P[0] = 0.0f;
#pragma unroll
  for (int d = 0; d < 10; ++d) P[d + 1] = P[d] + rowp[NHEAD + h * 10 + d];
  const float c10 = P[10];

  float* orow = out + ((size_t)b * NHEAD + h) * NQ;
  for (int j = j0; j < NQ; j += 4) {
    int r = inv[j];
    int delta = r - mi;
    int s = delta < 0 ? -delta : delta;
    float sign = (delta > 0) ? 1.0f : ((delta < 0) ? -1.0f : 0.0f);
    float off = (float)(s / 10) * c10 + P[s % 10];
    orow[j] = med + sign * off;
  }
}

// ---------------- launch ----------------
extern "C" void kernel_launch(void* const* d_in, const int* in_sizes, int n_in,
                              void* d_out, int out_size, void* d_ws, size_t ws_size,
                              hipStream_t stream) {
  const float* x  = (const float*)d_in[0];
  const float* qu = (const float*)d_in[1];
  const float* Wm = (const float*)d_in[2];
  const float* bm = (const float*)d_in[3];
  const float* Wi = (const float*)d_in[4];
  const float* bi = (const float*)d_in[5];
  float* out = (float*)d_out;

  const size_t nX = (size_t)B_ROWS * HID;   // 8,388,608
  const size_t nW = (size_t)NTOT * HID;     // 720,896

  __bf16* xh  = (__bf16*)d_ws;
  __bf16* xl  = xh + nX;
  __bf16* wth = xl + nX;
  __bf16* wtl = wth + nW;
  float*  pre = (float*)(wtl + nW);         // byte offset divisible by 4

  cvt_x_kernel<<<(unsigned)((nX + 255) / 256), 256, 0, stream>>>(x, xh, xl, (int)nX);
  cvt_w_kernel<<<(unsigned)((nW + 255) / 256), 256, 0, stream>>>(Wm, Wi, wth, wtl);

  dim3 ggrid(B_ROWS / 256, NTOT / 64);      // (32, 11)
  gemm_kernel<<<ggrid, 256, 0, stream>>>(xh, xl, wth, wtl, bm, bi, pre);

  quant_kernel<<<B_ROWS, 256, 0, stream>>>(qu, pre, out);
}
